// ChronoFormer_23802708754626
// MI455X (gfx1250) — compile-verified, matrix operand order checked by reading, not run on hardware
//
#include <hip/hip_runtime.h>
#include <hip/hip_bf16.h>

typedef __attribute__((ext_vector_type(16))) __bf16 v16bf;
typedef __attribute__((ext_vector_type(8)))  float  v8f;

#define D_   256
#define S_   2048
#define B_   4
#define H_   4
#define DK_  64
#define T_   1000
#define M_   (B_*S_)   /* 8192 rows */

union FA { v16bf v; unsigned u[8]; };
union FC { v8f   v; float    f[8]; };

// Load a 16x32 bf16 WMMA A/B fragment from row-major memory (global or LDS).
// Caller passes base = &mat[(tileRow + lane%16) * ld + k0].
// Per ISA layout: lanes 0-15 hold K = {0..7, 16..23}, lanes 16-31 hold K = {8..15, 24..31},
// packed 2 bf16 per VGPR -> uint indices {half*4 + j} and {half*4 + j + 4}.
__device__ __forceinline__ void load_frag(FA& fr, const __bf16* base, int half) {
  const unsigned* p = reinterpret_cast<const unsigned*>(base);
#pragma unroll
  for (int j = 0; j < 4; ++j) fr.u[j] = p[half * 4 + j];
#pragma unroll
  for (int j = 4; j < 8; ++j) fr.u[j] = p[half * 4 + j + 4];
}

__device__ __forceinline__ unsigned pk2(float a, float b) {
  union { __bf16 h; unsigned short u; } xa, xb;
  xa.h = (__bf16)a; xb.h = (__bf16)b;
  return (unsigned)xa.u | ((unsigned)xb.u << 16);
}

// ---------------------------------------------------------------- embeddings
__global__ void embed_kernel(const int* __restrict__ x, const int* __restrict__ td,
                             const float* __restrict__ ee, const float* __restrict__ te,
                             float* __restrict__ hf, __bf16* __restrict__ hb) {
  int row = blockIdx.x;
  int d   = threadIdx.x;
  int tb  = td[row]; tb = tb < 0 ? 0 : (tb > T_ - 1 ? T_ - 1 : tb);
  float v = ee[(size_t)x[row] * D_ + d] + te[(size_t)tb * D_ + d];
  size_t idx = (size_t)row * D_ + d;
  hf[idx] = v;
  hb[idx] = (__bf16)v;
}

// ---------------------------------------------- time-bias + mask -> additive bias
__global__ void bias_kernel(const int* __restrict__ td, const int* __restrict__ mask,
                            const float* __restrict__ tp, float* __restrict__ bm) {
  int idx = blockIdx.x * blockDim.x + threadIdx.x;
  if (idx >= B_ * S_) return;
  int bb = idx / S_, ss = idx % S_;
  int tb = td[idx]; tb = tb < 0 ? 0 : (tb > T_ - 1 ? T_ - 1 : tb);
  int m  = mask[idx];
#pragma unroll
  for (int h = 0; h < H_; ++h)
    bm[(size_t)(bb * H_ + h) * S_ + ss] = m ? tp[tb * H_ + h] : -1.0e9f;
}

// ------------------------------------- weight transpose + fp32 -> bf16 ([N][K])
__global__ void wt_kernel(const float* __restrict__ W, __bf16* __restrict__ Wt) {
  int kk = threadIdx.x;   // K index
  int n  = blockIdx.x;    // N index
  Wt[(size_t)n * D_ + kk] = (__bf16)W[(size_t)kk * D_ + n];
}

// --------------------------------------------------- GEMM: Y = A @ Wt^T + bias
// A: [M][K=256] bf16 row-major; Wt: [N=256][K=256] bf16; one 16x16 tile / wave.
// vmode 0: out row-major bf16 [M][256]; vmode 1: out transposed-V bf16 [B][H][DK][S].
__global__ void gemm_kernel(const __bf16* __restrict__ A, const __bf16* __restrict__ Wt,
                            const float* __restrict__ bias, __bf16* __restrict__ out,
                            float scale, int vmode) {
  int wave = threadIdx.x >> 5;
  int lane = threadIdx.x & 31;
  int half = lane >> 4, ln = lane & 15;
  int tile = blockIdx.x * (blockDim.x >> 5) + wave;
  const int NT = D_ / 16;
  int mt = tile / NT, nt = tile % NT;

  FC acc;
#pragma unroll
  for (int r = 0; r < 8; ++r) acc.f[r] = 0.0f;

  const __bf16* arow = A  + (size_t)(mt * 16 + ln) * D_;
  const __bf16* brow = Wt + (size_t)(nt * 16 + ln) * D_;
#pragma unroll
  for (int kk = 0; kk < D_; kk += 32) {
    FA a, b;
    load_frag(a, arow + kk, half);
    load_frag(b, brow + kk, half);
    acc.v = __builtin_amdgcn_wmma_f32_16x16x32_bf16(false, a.v, false, b.v,
                                                    (short)0, acc.v, false, false);
  }
  float bn = bias[nt * 16 + ln];
  if (vmode == 0) {
#pragma unroll
    for (int r = 0; r < 8; ++r) {
      int row = mt * 16 + half * 8 + r;
      out[(size_t)row * D_ + nt * 16 + ln] = (__bf16)((acc.f[r] + bn) * scale);
    }
  } else {
    int n  = nt * 16 + ln;
    int hh = n >> 6, dk = n & (DK_ - 1);
#pragma unroll
    for (int r = 0; r < 8; ++r) {
      int row = mt * 16 + half * 8 + r;
      int bb  = row >> 11, ss = row & (S_ - 1);
      out[((size_t)(bb * H_ + hh) * DK_ + dk) * S_ + ss] = (__bf16)((acc.f[r] + bn) * scale);
    }
  }
}

// ----------------------------------------------- flash attention (transposed)
// q,k: [B][S][D] bf16 (q pre-scaled by 1/8); vt: [B][H][DK][S] bf16;
// bm: [B][H][S] fp32 additive bias (mask folded). ctx out: [B][S][D] bf16.
// All 8 waves of a block share one (b,h): K/V tiles are staged once per block
// into LDS with CDNA5 async global->LDS copies (ASYNCcnt), then fragments are
// fed from LDS, removing the 8x redundant global streaming.
__global__ void __launch_bounds__(256) attn_kernel(
    const __bf16* __restrict__ q, const __bf16* __restrict__ k,
    const __bf16* __restrict__ vt, const float* __restrict__ bm,
    __bf16* __restrict__ ctx) {
  // Single shared buffer -> allocated at LDS offset 0.
  // [0      .. 4096) : K tile, 32 keys x 64 dims, row-major (128B rows)
  // [4096   .. 8192) : V^T tile, 64 dims x 32 keys, row-major (64B rows)
  __shared__ __bf16 smem[32 * 64 + 64 * 32];
  __bf16* lk = smem;
  __bf16* lv = smem + 32 * 64;

  int wave = threadIdx.x >> 5;
  int lane = threadIdx.x & 31;
  int half = lane >> 4, ln = lane & 15;
  int task = blockIdx.x * 8 + wave;
  int qt = task & 127;            // S/16 query tiles
  int hh = (task >> 7) & (H_ - 1);
  int bb = task >> 9;

  // Q^T as B-operand fragments (lane = query, VGPR = head-dim pairs)
  const __bf16* qrow = q + (size_t)(bb * S_ + qt * 16 + ln) * D_ + hh * DK_;
  FA qf0, qf1;
  load_frag(qf0, qrow,      half);
  load_frag(qf1, qrow + 32, half);

  float mrun = -1.0e30f, lrun = 0.0f;
  FC o[4];
#pragma unroll
  for (int dc = 0; dc < 4; ++dc)
#pragma unroll
    for (int r = 0; r < 8; ++r) o[dc].f[r] = 0.0f;

  const float*  brow    = bm + (size_t)(bb * H_ + hh) * S_;
  const __bf16* kbase_g = k  + (size_t)bb * S_ * D_ + hh * DK_;
  const __bf16* vbase_g = vt + (size_t)(bb * H_ + hh) * DK_ * S_;

  // Per-thread async-copy assignment: 256 threads x 16B = 4KB per tile.
  int t = threadIdx.x;
  int krow = t >> 3, kch = t & 7;      // K: 32 rows x 8 chunks
  int vrow = t >> 2, vch = t & 3;      // V: 64 rows x 4 chunks
  unsigned lkoff = (unsigned)((krow * 64 + kch * 8) * 2);            // LDS byte addr
  unsigned lvoff = (unsigned)(32 * 64 * 2 + (vrow * 32 + vch * 8) * 2);

  for (int kt = 0; kt < S_; kt += 32) {
    __syncthreads();   // previous iteration's LDS reads complete before overwrite

    const __bf16* gk = kbase_g + (size_t)(kt + krow) * D_ + kch * 8;
    const __bf16* gv = vbase_g + (size_t)vrow * S_ + kt + vch * 8;
    asm volatile("global_load_async_to_lds_b128 %0, %1, off"
                 :: "v"(lkoff), "v"(gk) : "memory");
    asm volatile("global_load_async_to_lds_b128 %0, %1, off"
                 :: "v"(lvoff), "v"(gv) : "memory");
    if (kt + 32 < S_)
      __builtin_prefetch(kbase_g + (size_t)(kt + 32 + krow) * D_, 0, 1);
    asm volatile("s_wait_asynccnt 0" ::: "memory");
    __syncthreads();   // tiles visible to all waves

    // S^T = K * Q^T : C-layout -> lane = query, VGPR r = key (r + half*8)
    FC s0, s1;
    {
      const __bf16* krw = lk + (size_t)ln * 64;
      FA ka0, ka1; load_frag(ka0, krw, half); load_frag(ka1, krw + 32, half);
      FC z;
#pragma unroll
      for (int r = 0; r < 8; ++r) z.f[r] = 0.0f;
      z.v = __builtin_amdgcn_wmma_f32_16x16x32_bf16(false, ka0.v, false, qf0.v, (short)0, z.v, false, false);
      z.v = __builtin_amdgcn_wmma_f32_16x16x32_bf16(false, ka1.v, false, qf1.v, (short)0, z.v, false, false);
      s0 = z;
    }
    {
      const __bf16* krw = lk + (size_t)(16 + ln) * 64;
      FA ka0, ka1; load_frag(ka0, krw, half); load_frag(ka1, krw + 32, half);
      FC z;
#pragma unroll
      for (int r = 0; r < 8; ++r) z.f[r] = 0.0f;
      z.v = __builtin_amdgcn_wmma_f32_16x16x32_bf16(false, ka0.v, false, qf0.v, (short)0, z.v, false, false);
      z.v = __builtin_amdgcn_wmma_f32_16x16x32_bf16(false, ka1.v, false, qf1.v, (short)0, z.v, false, false);
      s1 = z;
    }

    // per-key bias (+mask) — key index = kt + sub*16 + half*8 + r
#pragma unroll
    for (int r = 0; r < 8; ++r) {
      s0.f[r] += brow[kt +      half * 8 + r];
      s1.f[r] += brow[kt + 16 + half * 8 + r];
    }

    // online softmax over keys; stats are per query (= per lane)
    float tmax = -1.0e30f;
#pragma unroll
    for (int r = 0; r < 8; ++r) tmax = fmaxf(tmax, fmaxf(s0.f[r], s1.f[r]));
    tmax = fmaxf(tmax, __shfl_xor(tmax, 16, 32));
    float mnew  = fmaxf(mrun, tmax);
    float alpha = __expf(mrun - mnew);
    float tsum  = 0.0f;
#pragma unroll
    for (int r = 0; r < 8; ++r) {
      s0.f[r] = __expf(s0.f[r] - mnew); tsum += s0.f[r];
      s1.f[r] = __expf(s1.f[r] - mnew); tsum += s1.f[r];
    }
    tsum += __shfl_xor(tsum, 16, 32);
    lrun = lrun * alpha + tsum;
    mrun = mnew;

    // Repack P^T into a B-fragment: the C-layout key indices per lane exactly
    // match the B-layout K mapping -> pure per-lane bf16 packing, no shuffles.
    FA pf;
#pragma unroll
    for (int j = 0; j < 4; ++j) {
      pf.u[j]     = pk2(s0.f[2 * j], s0.f[2 * j + 1]);
      pf.u[4 + j] = pk2(s1.f[2 * j], s1.f[2 * j + 1]);
    }

    // O^T = alpha * O^T + V^T * P^T  (4 chunks of 16 head-dims) — V from LDS
#pragma unroll
    for (int dc = 0; dc < 4; ++dc) {
      FA vf;
      load_frag(vf, lv + (size_t)(dc * 16 + ln) * 32, half);
#pragma unroll
      for (int r = 0; r < 8; ++r) o[dc].f[r] *= alpha;
      o[dc].v = __builtin_amdgcn_wmma_f32_16x16x32_bf16(false, vf.v, false, pf.v,
                                                        (short)0, o[dc].v, false, false);
    }
  }

  float inv = 1.0f / lrun;
  // O^T C-layout: lane = query, d = dc*16 + half*8 + r  -> 16B aligned stores
  __bf16* crow = ctx + (size_t)(bb * S_ + qt * 16 + ln) * D_ + hh * DK_ + half * 8;
#pragma unroll
  for (int dc = 0; dc < 4; ++dc) {
    unsigned pk[4];
#pragma unroll
    for (int j = 0; j < 4; ++j)
      pk[j] = pk2(o[dc].f[2 * j] * inv, o[dc].f[2 * j + 1] * inv);
    uint4 st; st.x = pk[0]; st.y = pk[1]; st.z = pk[2]; st.w = pk[3];
    *reinterpret_cast<uint4*>(crow + dc * 16) = st;
  }
}

// -------------------------------- output projection fused with LayerNorm
// block = 512 threads = 16 waves; wave w -> column tile w; block -> 16 rows.
__global__ void proj_ln_kernel(const __bf16* __restrict__ A, const __bf16* __restrict__ Wt,
                               const float* __restrict__ bias, const float* __restrict__ g,
                               const float* __restrict__ be, float* __restrict__ hf,
                               __bf16* __restrict__ hb) {
  __shared__ float ssum[16], ssq[16];
  int wave = threadIdx.x >> 5;
  int lane = threadIdx.x & 31;
  int half = lane >> 4, ln = lane & 15;
  int mt = blockIdx.x;
  int nt = wave;
  if (threadIdx.x < 16) { ssum[threadIdx.x] = 0.0f; ssq[threadIdx.x] = 0.0f; }
  __syncthreads();

  FC acc;
#pragma unroll
  for (int r = 0; r < 8; ++r) acc.f[r] = 0.0f;
  const __bf16* arow = A  + (size_t)(mt * 16 + ln) * D_;
  const __bf16* brow = Wt + (size_t)(nt * 16 + ln) * D_;
#pragma unroll
  for (int kk = 0; kk < D_; kk += 32) {
    FA a, b;
    load_frag(a, arow + kk, half);
    load_frag(b, brow + kk, half);
    acc.v = __builtin_amdgcn_wmma_f32_16x16x32_bf16(false, a.v, false, b.v,
                                                    (short)0, acc.v, false, false);
  }
  float bn = bias[nt * 16 + ln];
#pragma unroll
  for (int r = 0; r < 8; ++r) acc.f[r] += bn;

  // row statistics: shuffle-reduce across the 16 column lanes, LDS-atomics across waves
#pragma unroll
  for (int r = 0; r < 8; ++r) {
    float v = acc.f[r], v2 = v * v;
#pragma unroll
    for (int d = 1; d < 16; d <<= 1) {
      v  += __shfl_xor(v,  d, 32);
      v2 += __shfl_xor(v2, d, 32);
    }
    if (ln == 0) {
      atomicAdd(&ssum[half * 8 + r], v);
      atomicAdd(&ssq [half * 8 + r], v2);
    }
  }
  __syncthreads();

#pragma unroll
  for (int r = 0; r < 8; ++r) {
    int   row  = half * 8 + r;
    float mean = ssum[row] * (1.0f / D_);
    float var  = ssq [row] * (1.0f / D_) - mean * mean;
    float rstd = rsqrtf(var + 1e-5f);
    int   col  = nt * 16 + ln;
    float y    = (acc.f[r] - mean) * rstd * g[col] + be[col];
    size_t idx = (size_t)(mt * 16 + row) * D_ + col;
    hf[idx] = y;
    hb[idx] = (__bf16)y;
  }
}

// ------------------------------------------------------------- classifier head
__global__ void head_kernel(const float* __restrict__ hf, const float* __restrict__ Wc1,
                            const float* __restrict__ bc1, const float* __restrict__ Wc2,
                            const float* __restrict__ bc2, float* __restrict__ out) {
  __shared__ float red[128];
  int j = threadIdx.x;
  for (int b = 0; b < B_; ++b) {
    const float* last = hf + (size_t)(b * S_ + S_ - 1) * D_;
    float acc = bc1[j];
    for (int kk = 0; kk < D_; ++kk) acc += last[kk] * Wc1[kk * 128 + j];
    float hid = fmaxf(acc, 0.0f);
    red[j] = hid * Wc2[j];
    __syncthreads();
    for (int st = 64; st > 0; st >>= 1) {
      if (j < st) red[j] += red[j + st];
      __syncthreads();
    }
    if (j == 0) out[b] = 1.0f / (1.0f + __expf(-(red[0] + bc2[0])));
    __syncthreads();
  }
}

// ------------------------------------------------------------------ launcher
extern "C" void kernel_launch(void* const* d_in, const int* in_sizes, int n_in,
                              void* d_out, int out_size, void* d_ws, size_t ws_size,
                              hipStream_t stream) {
  const int*   x    = (const int*)  d_in[0];
  const int*   td   = (const int*)  d_in[1];
  const int*   mask = (const int*)  d_in[2];
  const float* ee   = (const float*)d_in[3];
  const float* te   = (const float*)d_in[4];
  const float* Wq   = (const float*)d_in[5];
  const float* bq   = (const float*)d_in[6];
  const float* Wk   = (const float*)d_in[7];
  const float* bk   = (const float*)d_in[8];
  const float* Wv   = (const float*)d_in[9];
  const float* bv   = (const float*)d_in[10];
  const float* tp   = (const float*)d_in[11];
  const float* Wo   = (const float*)d_in[12];
  const float* bo   = (const float*)d_in[13];
  const float* lg   = (const float*)d_in[14];
  const float* lb   = (const float*)d_in[15];
  const float* Wc1  = (const float*)d_in[16];
  const float* bc1  = (const float*)d_in[17];
  const float* Wc2  = (const float*)d_in[18];
  const float* bc2  = (const float*)d_in[19];

  char*  ws  = (char*)d_ws;
  size_t off = 0;
  auto carve = [&](size_t bytes) -> char* {
    char* p = ws + off;
    off += (bytes + 255) & ~(size_t)255;
    return p;
  };

  float*  hf   = (float*) carve((size_t)M_ * D_ * 4);
  __bf16* hb   = (__bf16*)carve((size_t)M_ * D_ * 2);
  __bf16* qb   = (__bf16*)carve((size_t)M_ * D_ * 2);
  __bf16* kb   = (__bf16*)carve((size_t)M_ * D_ * 2);
  __bf16* vtb  = (__bf16*)carve((size_t)M_ * D_ * 2);
  __bf16* ctxb = (__bf16*)carve((size_t)M_ * D_ * 2);
  float*  bm   = (float*) carve((size_t)B_ * H_ * S_ * 4);
  __bf16* wqt  = (__bf16*)carve((size_t)D_ * D_ * 2);
  __bf16* wkt  = (__bf16*)carve((size_t)D_ * D_ * 2);
  __bf16* wvt  = (__bf16*)carve((size_t)D_ * D_ * 2);
  __bf16* wot  = (__bf16*)carve((size_t)D_ * D_ * 2);

  embed_kernel<<<M_, D_, 0, stream>>>(x, td, ee, te, hf, hb);

  const int GEMM_BLOCKS = (M_ / 16) * (D_ / 16) / 8;   // 8 waves / block
  const int ATTN_BLOCKS = (B_ * H_ * (S_ / 16)) / 8;

  for (int l = 0; l < 2; ++l) {
    size_t wo = (size_t)l * D_ * D_;
    wt_kernel<<<D_, D_, 0, stream>>>(Wq + wo, wqt);
    wt_kernel<<<D_, D_, 0, stream>>>(Wk + wo, wkt);
    wt_kernel<<<D_, D_, 0, stream>>>(Wv + wo, wvt);
    wt_kernel<<<D_, D_, 0, stream>>>(Wo + wo, wot);
    bias_kernel<<<(B_ * S_ + 255) / 256, 256, 0, stream>>>(td, mask, tp + (size_t)l * T_ * H_, bm);

    gemm_kernel<<<GEMM_BLOCKS, 256, 0, stream>>>(hb, wqt, bq + l * D_, qb, 0.125f, 0);
    gemm_kernel<<<GEMM_BLOCKS, 256, 0, stream>>>(hb, wkt, bk + l * D_, kb, 1.0f, 0);
    gemm_kernel<<<GEMM_BLOCKS, 256, 0, stream>>>(hb, wvt, bv + l * D_, vtb, 1.0f, 1);

    attn_kernel<<<ATTN_BLOCKS, 256, 0, stream>>>(qb, kb, vtb, bm, ctxb);

    proj_ln_kernel<<<M_ / 16, 512, 0, stream>>>(ctxb, wot, bo + l * D_, lg + l * D_, lb + l * D_, hf, hb);
  }

  head_kernel<<<1, 128, 0, stream>>>(hf, Wc1, bc1, Wc2, bc2, (float*)d_out);
}
